// GATCell_33517924778634
// MI455X (gfx1250) — compile-verified, matrix operand order checked by reading, not run on hardware
//
#include <hip/hip_runtime.h>

// GAT cell, 2 layers, B=16 S=1024 D=768.
// Per layer: cvt(X->f16) -> tiled-WMMA linear -> scores -> col-softmax stats ->
//            attn weights (f16) -> tiled-WMMA aggregation -> d_out slice.
// Both GEMMs: 256-thread block computes 128x128, LDS double-buffered 128x32
// operand tiles staged with GLOBAL_LOAD_ASYNC_TO_LDS_B128 (fallback: sync copy),
// 8 v_wmma_f32_16x16x32_f16 per K-step per wave.

#define B_    16
#define S_    1024
#define DIN   768
#define DOUT  768
#define ALPHA 0.2f
#define NEGINF (-1.0e12f)
#define LDSP  40   // padded LDS tile pitch in halves (80B) to spread banks
#define KT    32   // K-tile

typedef __attribute__((ext_vector_type(16))) _Float16 v16h;
typedef __attribute__((ext_vector_type(8)))  _Float16 v8h;
typedef __attribute__((ext_vector_type(4)))  _Float16 v4h;
typedef __attribute__((ext_vector_type(8)))  float    v8f;
typedef __attribute__((ext_vector_type(4)))  float    v4f;
typedef __attribute__((ext_vector_type(4)))  int      v4i;

// AS-qualified pointee types for the async-copy builtin (b128 = int4)
typedef __attribute__((address_space(1))) v4i gv4i;
typedef __attribute__((address_space(3))) v4i lv4i;

#if defined(__has_builtin)
#if __has_builtin(__builtin_amdgcn_global_load_async_to_lds_b128)
#define HAS_ASYNC_LDS 1
#endif
#endif
#ifndef HAS_ASYNC_LDS
#define HAS_ASYNC_LDS 0
#endif

__device__ __forceinline__ v8f wmma_f16(v16h a, v16h b, v8f c) {
  return __builtin_amdgcn_wmma_f32_16x16x32_f16(false, a, false, b, (short)0, c,
                                                false, false);
}

// one 16B chunk global -> LDS (async if available)
__device__ __forceinline__ void cp_b128(const _Float16* __restrict__ g,
                                        _Float16* __restrict__ l) {
#if HAS_ASYNC_LDS
  __builtin_amdgcn_global_load_async_to_lds_b128((gv4i*)g, (lv4i*)l, 0, 0);
#else
  *(v8h*)l = *(const v8h*)g;
#endif
}

__device__ __forceinline__ void wait_stage() {
#if HAS_ASYNC_LDS
#if __has_builtin(__builtin_amdgcn_s_wait_asynccnt)
  __builtin_amdgcn_s_wait_asynccnt(0);
#else
  asm volatile("s_wait_asynccnt 0x0" ::: "memory");
#endif
#endif
}

// stage a 128-row x 32-half tile (rows at gbase + row*ld + k0) into lbuf (pitch LDSP)
// straight-line: 2 chunks per thread, second chunk = row + 64 (constant offsets)
__device__ __forceinline__ void stage_tile(const _Float16* __restrict__ gbase,
                                           size_t ld, int k0,
                                           _Float16* __restrict__ lbuf, int tid) {
  const int row = tid >> 2;                 // 0..63
  const int ko  = (tid & 3) * 8;            // 0,8,16,24 halves
  const _Float16* g = gbase + (size_t)row * ld + (size_t)(k0 + ko);
  _Float16* l = lbuf + row * LDSP + ko;
  cp_b128(g, l);
  cp_b128(g + 64 * ld, l + 64 * LDSP);      // rows 64..127
}

// WMMA f16 A(16x32)/B(32x16) fragment from an LDS tile row (row already applied).
// lane holds row/col (lane&15); sel=(lane>>4): chunks {sel*8..+7, 16+sel*8..+7}.
__device__ __forceinline__ v16h frag_lds(const _Float16* __restrict__ p, int sel) {
  v8h a = *(const v8h*)(p + sel * 8);
  v8h b = *(const v8h*)(p + 16 + sel * 8);
  v16h f;
#pragma unroll
  for (int i = 0; i < 8; ++i) { f[i] = a[i]; f[8 + i] = b[i]; }
  return f;
}

// ---------------- f32 -> f16 pre-convert (vectorized) ------------------------------
__global__ void cvt_f32_f16(const float* __restrict__ src, _Float16* __restrict__ dst,
                            int n4) {
  const int i = blockIdx.x * blockDim.x + threadIdx.x;
  if (i < n4) {
    v4f x = ((const v4f*)src)[i];
    v4h o;
#pragma unroll
    for (int v = 0; v < 4; ++v) o[v] = (_Float16)x[v];
    ((v4h*)dst)[i] = o;
  }
}

// ---------------- 1) feature linear: hT = f16(X @ W^T + b), stored transposed ------
// grid (DOUT/128, B*S/128), block 256 (8 waves as 4(M) x 2(N); wave => 32x64).
__global__ void __launch_bounds__(256)
gat_linear(const _Float16* __restrict__ Xh, const _Float16* __restrict__ Wh,
           const float* __restrict__ Wb, _Float16* __restrict__ hT) {
  __shared__ __align__(16) _Float16 sA[2][128 * LDSP];
  __shared__ __align__(16) _Float16 sB[2][128 * LDSP];
  const int tid  = threadIdx.x;
  const int lane = tid & 31, wave = tid >> 5;
  const int r = lane & 15, sel = lane >> 4;
  const int wy = wave >> 1, wx = wave & 1;
  const int row0 = blockIdx.y * 128;              // global row over (b,s)
  const int col0 = blockIdx.x * 128;              // output feature
  const _Float16* Abase = Xh + (size_t)row0 * DIN;
  const _Float16* Bbase = Wh + (size_t)col0 * DIN;

  const v8f zero = {0.f, 0.f, 0.f, 0.f, 0.f, 0.f, 0.f, 0.f};
  v8f acc[8];
#pragma unroll
  for (int t = 0; t < 8; ++t) acc[t] = zero;

  stage_tile(Abase, DIN, 0, sA[0], tid);
  stage_tile(Bbase, DIN, 0, sB[0], tid);
  wait_stage();
  __syncthreads();

  const int NK = DIN / KT;                        // 24
  for (int ks = 0; ks < NK; ++ks) {
    const int cur = ks & 1, nxt = cur ^ 1;
    if (ks + 1 < NK) {                            // prefetch next K-tile into LDS
      stage_tile(Abase, DIN, (ks + 1) * KT, sA[nxt], tid);
      stage_tile(Bbase, DIN, (ks + 1) * KT, sB[nxt], tid);
    }
    v16h af[2], bf[4];
#pragma unroll
    for (int im = 0; im < 2; ++im)
      af[im] = frag_lds(&sA[cur][(wy * 32 + im * 16 + r) * LDSP], sel);
#pragma unroll
    for (int in = 0; in < 4; ++in)
      bf[in] = frag_lds(&sB[cur][(wx * 64 + in * 16 + r) * LDSP], sel);
#pragma unroll
    for (int im = 0; im < 2; ++im)
#pragma unroll
      for (int in = 0; in < 4; ++in)
        acc[im * 4 + in] = wmma_f16(af[im], bf[in], acc[im * 4 + in]);
    wait_stage();
    __syncthreads();
  }

  // C/D layout: VGPR v -> M = v + 8*sel, N = r. Store transposed, 8 contiguous halves.
  const int b  = row0 >> 10;
  const int s0 = row0 & (S_ - 1);
#pragma unroll
  for (int im = 0; im < 2; ++im) {
    const int sm = s0 + wy * 32 + im * 16 + 8 * sel;
#pragma unroll
    for (int in = 0; in < 4; ++in) {
      const int col  = col0 + wx * 64 + in * 16 + r;
      const float bi = Wb[col];
      v8h o;
#pragma unroll
      for (int v = 0; v < 8; ++v) o[v] = (_Float16)(acc[im * 4 + in][v] + bi);
      *(v8h*)(hT + (((size_t)b * DOUT + col) << 10) + sm) = o;
    }
  }
}

// ---------------- 2) attention scores s_i, s_j per (b,s) ---------------------------
__global__ void gat_scores(const _Float16* __restrict__ hT, const float* __restrict__ A,
                           float* __restrict__ s_i, float* __restrict__ s_j) {
  const int gid = blockIdx.x * blockDim.x + threadIdx.x;  // 0 .. B*S-1
  const int b = gid >> 10, s = gid & (S_ - 1);
  const _Float16* hb = hT + ((size_t)b * DOUT << 10) + s;
  const float* a_src = A;
  const float* a_dst = A + DOUT;
  float si = 0.f, sj = 0.f;
  for (int o = 0; o < DOUT; ++o) {                // coalesced over s
    float hv = (float)hb[(size_t)o << 10];
    si = fmaf(hv, a_src[o], si);
    sj = fmaf(hv, a_dst[o], sj);
  }
  s_i[gid] = si;
  s_j[gid] = sj;
}

// ---------------- 3) column softmax stats (softmax over axis=1 => over i) ----------
__global__ void gat_colstats(const int* __restrict__ mask, const float* __restrict__ s_i,
                             const float* __restrict__ s_j, const float* __restrict__ Ab,
                             float* __restrict__ cmax, float* __restrict__ crcp) {
  const int gid = blockIdx.x * blockDim.x + threadIdx.x;  // (b, j)
  const int b = gid >> 10, j = gid & (S_ - 1);
  const float ab = Ab[0];
  const float sj = s_j[gid];
  const int* mcol  = mask + ((size_t)b << 20) + j;
  const float* sib = s_i + (b << 10);
  float m = NEGINF, sum = 0.f;
  for (int i = 0; i < S_; ++i) {                  // coalesced over j for each i
    float x = sib[i] + sj + ab;
    float e = (mcol[(size_t)i << 10] > 0) ? (x > 0.f ? x : ALPHA * x) : NEGINF;
    float mn = fmaxf(m, e);
    sum = sum * __expf(m - mn) + __expf(e - mn);
    m = mn;
  }
  cmax[gid] = m;
  crcp[gid] = 1.0f / sum;
}

// ---------------- 4) attention weights, f16 ----------------------------------------
__global__ void gat_attn(const int* __restrict__ mask, const float* __restrict__ s_i,
                         const float* __restrict__ s_j, const float* __restrict__ Ab,
                         const float* __restrict__ cmax, const float* __restrict__ crcp,
                         _Float16* __restrict__ attn) {
  const size_t gid = (size_t)blockIdx.x * blockDim.x + threadIdx.x;  // (b, i, j)
  const int j  = (int)(gid & (S_ - 1));
  const int bi = (int)(gid >> 10);
  const int b  = bi >> 10;
  const int bj = (b << 10) + j;
  float x = s_i[bi] + s_j[bj] + Ab[0];
  float e = x > 0.f ? x : ALPHA * x;
  float w = 0.f;
  if (mask[gid] > 0) w = __expf(e - cmax[bj]) * crcp[bj];
  attn[gid] = (_Float16)w;
}

// ---------------- 5) aggregation: out = attn @ h -----------------------------------
// A = attn rows (pitch S); B^T = hT rows (column d of h, contiguous over j).
__global__ void __launch_bounds__(256)
gat_agg(const _Float16* __restrict__ attn, const _Float16* __restrict__ hT,
        float* __restrict__ out) {
  __shared__ __align__(16) _Float16 sA[2][128 * LDSP];
  __shared__ __align__(16) _Float16 sB[2][128 * LDSP];
  const int tid  = threadIdx.x;
  const int lane = tid & 31, wave = tid >> 5;
  const int r = lane & 15, sel = lane >> 4;
  const int wy = wave >> 1, wx = wave & 1;
  const int row0 = blockIdx.y * 128;              // global row over (b,i); 128 | S
  const int col0 = blockIdx.x * 128;              // output feature d
  const int b = row0 >> 10;
  const _Float16* Abase = attn + ((size_t)row0 << 10);
  const _Float16* Bbase = hT + (((size_t)b * DOUT + col0) << 10);

  const v8f zero = {0.f, 0.f, 0.f, 0.f, 0.f, 0.f, 0.f, 0.f};
  v8f acc[8];
#pragma unroll
  for (int t = 0; t < 8; ++t) acc[t] = zero;

  stage_tile(Abase, S_, 0, sA[0], tid);
  stage_tile(Bbase, S_, 0, sB[0], tid);
  wait_stage();
  __syncthreads();

  const int NK = S_ / KT;                         // 32
  for (int ks = 0; ks < NK; ++ks) {
    const int cur = ks & 1, nxt = cur ^ 1;
    if (ks + 1 < NK) {
      stage_tile(Abase, S_, (ks + 1) * KT, sA[nxt], tid);
      stage_tile(Bbase, S_, (ks + 1) * KT, sB[nxt], tid);
    }
    v16h af[2], bf[4];
#pragma unroll
    for (int im = 0; im < 2; ++im)
      af[im] = frag_lds(&sA[cur][(wy * 32 + im * 16 + r) * LDSP], sel);
#pragma unroll
    for (int in = 0; in < 4; ++in)
      bf[in] = frag_lds(&sB[cur][(wx * 64 + in * 16 + r) * LDSP], sel);
#pragma unroll
    for (int im = 0; im < 2; ++im)
#pragma unroll
      for (int in = 0; in < 4; ++in)
        acc[im * 4 + in] = wmma_f16(af[im], bf[in], acc[im * 4 + in]);
    wait_stage();
    __syncthreads();
  }

#pragma unroll
  for (int im = 0; im < 2; ++im) {
    const int rowm = row0 + wy * 32 + im * 16 + 8 * sel;
#pragma unroll
    for (int in = 0; in < 4; ++in) {
      const int col = col0 + wx * 64 + in * 16 + r;
#pragma unroll
      for (int v = 0; v < 8; ++v)
        out[(size_t)(rowm + v) * DOUT + col] = acc[im * 4 + in][v];
    }
  }
}

extern "C" void kernel_launch(void* const* d_in, const int* in_sizes, int n_in,
                              void* d_out, int out_size, void* d_ws, size_t ws_size,
                              hipStream_t stream) {
  (void)in_sizes; (void)n_in; (void)out_size; (void)ws_size;
  const float* input = (const float*)d_in[0];        // (B,S,DIN) f32
  const int*   mask  = (const int*)d_in[1];          // (B,S,S) i32
  const float* W     = (const float*)d_in[2];        // (L,DOUT,DIN)
  const float* Wb    = (const float*)d_in[3];        // (L,DOUT)
  const float* A     = (const float*)d_in[4];        // (L,2*DOUT)
  const float* Ab    = (const float*)d_in[5];        // (L,)
  float* out = (float*)d_out;                        // (L,B,S,DOUT)

  // workspace: Xh | Wh(2 layers) | hT | attn | stats
  _Float16* Xh   = (_Float16*)d_ws;
  _Float16* Wh   = Xh + (size_t)B_ * S_ * DIN;
  _Float16* hT   = Wh + (size_t)2 * DOUT * DIN;
  _Float16* attn = hT + (size_t)B_ * DOUT * S_;
  float* s_i = (float*)(attn + (size_t)B_ * S_ * S_);
  float* s_j = s_i + B_ * S_;
  float* cmx = s_j + B_ * S_;
  float* crp = cmx + B_ * S_;

  const dim3 gemm_grid(DOUT / 128, (B_ * S_) / 128);
  const dim3 gemm_blk(256);
  const size_t BSD = (size_t)B_ * S_ * DOUT;

  // convert both layers' W to f16 once per call
  {
    const int n4 = 2 * DOUT * DIN / 4;
    cvt_f32_f16<<<dim3((n4 + 255) / 256), dim3(256), 0, stream>>>(W, Wh, n4);
  }

  for (int l = 0; l < 2; ++l) {
    const float* X = (l == 0) ? input : (out + (size_t)(l - 1) * BSD);
    {
      const int n4 = (int)(BSD / 4);
      cvt_f32_f16<<<dim3((n4 + 255) / 256), dim3(256), 0, stream>>>(X, Xh, n4);
    }
    gat_linear<<<gemm_grid, gemm_blk, 0, stream>>>(
        Xh, Wh + (size_t)l * DOUT * DIN, Wb + (size_t)l * DOUT, hT);
    gat_scores<<<dim3(B_ * S_ / 256), dim3(256), 0, stream>>>(
        hT, A + (size_t)l * 2 * DOUT, s_i, s_j);
    gat_colstats<<<dim3(B_ * S_ / 256), dim3(256), 0, stream>>>(
        mask, s_i, s_j, Ab + l, cmx, crp);
    gat_attn<<<dim3((unsigned)(((size_t)B_ * S_ * S_) / 256)), dim3(256), 0, stream>>>(
        mask, s_i, s_j, Ab + l, cmx, crp, attn);
    gat_agg<<<gemm_grid, gemm_blk, 0, stream>>>(attn, hT, out + (size_t)l * BSD);
  }
}
